// LSTM_22763326669221
// MI455X (gfx1250) — compile-verified, matrix operand order checked by reading, not run on hardware
//
#include <hip/hip_runtime.h>

typedef _Float16 v16h __attribute__((ext_vector_type(16)));
typedef _Float16 v8h  __attribute__((ext_vector_type(8)));
typedef float    v8f  __attribute__((ext_vector_type(8)));

#define SEQ   3
#define BATCH 524288
#define INP   2
#define HID   32
#define NCLS  2
#define FITER 3
#define NT    8                // 4*HID/16 gate-column tiles
#define WAVES_PER_BLOCK 8
#define ROWS_PER_WAVE   16

__device__ __forceinline__ float fast_sigmoid(float x) {
    return __builtin_amdgcn_rcpf(1.0f + __expf(-x));
}
__device__ __forceinline__ float fast_tanh(float x) {
    // tanh(x) = 1 - 2/(exp(2x)+1)
    return 1.0f - 2.0f * __builtin_amdgcn_rcpf(__expf(2.0f * x) + 1.0f);
}

__global__ __launch_bounds__(256) void lstm_wmma_kernel(
    const float* __restrict__ x,
    const float* __restrict__ W_ih,
    const float* __restrict__ W_hh,
    const float* __restrict__ b_ih,
    const float* __restrict__ b_hh,
    const float* __restrict__ W_fc,
    const float* __restrict__ b_fc,
    float* __restrict__ out)
{
    // per-wave LDS: h transpose buffer (16x32 f16) + x window (3x16x2 f32)
    __shared__ __attribute__((aligned(16))) _Float16 hbuf_all[WAVES_PER_BLOCK][ROWS_PER_WAVE * HID];
    __shared__ __attribute__((aligned(16))) float    xbuf_all[WAVES_PER_BLOCK][SEQ][ROWS_PER_WAVE][INP];

    const int lane = threadIdx.x & 31;
    const int wave = threadIdx.x >> 5;
    const int b0   = (blockIdx.x * WAVES_PER_BLOCK + wave) * ROWS_PER_WAVE;

    _Float16* hbuf = hbuf_all[wave];
    float (*xbuf)[ROWS_PER_WAVE][INP] = xbuf_all[wave];

    const int j  = lane & 15;   // tile column (B/C/D) or A row
    const int hi = lane >> 4;   // half-wave select

    // ---- stage x window [3][16][2] into LDS ----
    if (lane < ROWS_PER_WAVE) {
        for (int t = 0; t < SEQ; ++t) {
            const float2 v = *(const float2*)&x[((size_t)t * BATCH + b0 + lane) * INP];
            xbuf[t][lane][0] = v.x;
            xbuf[t][lane][1] = v.y;
        }
    }

    // ---- preload W_hh as f16 B-tiles (K-striped like the A layout) ----
    // lane j holds column j of the 32x16 B tile; lanes 0-15: K 0..7,16..23;
    // lanes 16-31: K 8..15,24..31.  B[k][j] = W_hh[16t+j][k]
    v16h Bhh[NT];
    for (int t = 0; t < NT; ++t) {
        const float* wrow = &W_hh[(t * 16 + j) * HID];
        const int k0 = hi * 8;
        v16h b;
        #pragma unroll
        for (int i = 0; i < 8; ++i) b[i]     = (_Float16)wrow[k0 + i];
        #pragma unroll
        for (int i = 0; i < 8; ++i) b[8 + i] = (_Float16)wrow[16 + k0 + i];
        Bhh[t] = b;
    }
    // W_fc as a zero-padded B tile: column j = W_fc[j][*] for j < NCLS
    v16h Bfc;
    {
        const int k0 = hi * 8;
        #pragma unroll
        for (int i = 0; i < 16; ++i) Bfc[i] = (_Float16)0.0f;
        if (j < NCLS) {
            const float* wrow = &W_fc[j * HID];
            #pragma unroll
            for (int i = 0; i < 8; ++i) Bfc[i]     = (_Float16)wrow[k0 + i];
            #pragma unroll
            for (int i = 0; i < 8; ++i) Bfc[8 + i] = (_Float16)wrow[16 + k0 + i];
        }
    }

    // per-tile input projection params: n = 16t + j
    float wih0[NT], wih1[NT], bias[NT];
    for (int t = 0; t < NT; ++t) {
        const int n = t * 16 + j;
        wih0[t] = W_ih[n * INP + 0];
        wih1[t] = W_ih[n * INP + 1];
        bias[t] = b_ih[n] + b_hh[n];
    }
    const float fcbias = (j < NCLS) ? b_fc[j] : 0.0f;

    // ---- state: c in D layout (two col-half tiles), h in A layout (f16) ----
    v8f  c0 = {}, c1 = {};
    v16h hA = {};

    asm volatile("s_wait_dscnt 0" ::: "memory");   // x window visible wave-wide

    for (int step = 0; step < FITER * SEQ; ++step) {
        const int slot = step % SEQ;

        // per-row x values for this lane's D rows (row = r + 8*hi)
        float x0[8], x1[8];
        #pragma unroll
        for (int r = 0; r < 8; ++r) {
            const int row = r + 8 * hi;
            x0[r] = xbuf[slot][row][0];
            x1[r] = xbuf[slot][row][1];
        }

        // gates for hid-col halves 0..15 and 16..31
        #pragma unroll
        for (int half = 0; half < 2; ++half) {
            v8f g[4];   // i, f, g, o tiles for this column half
            #pragma unroll
            for (int gi = 0; gi < 4; ++gi) {
                const int t = gi * 2 + half;
                v8f acc;
                #pragma unroll
                for (int r = 0; r < 8; ++r)
                    acc[r] = fmaf(x0[r], wih0[t], fmaf(x1[r], wih1[t], bias[t]));
                g[gi] = __builtin_amdgcn_wmma_f32_16x16x32_f16(
                    false, hA, false, Bhh[t], (short)0, acc, false, false);
            }
            v8f& c = half ? c1 : c0;
            v8f hD;
            #pragma unroll
            for (int r = 0; r < 8; ++r) {
                const float ig = fast_sigmoid(g[0][r]);
                const float fg = fast_sigmoid(g[1][r]);
                const float gg = fast_tanh   (g[2][r]);
                const float og = fast_sigmoid(g[3][r]);
                const float cv = fmaf(fg, c[r], ig * gg);
                c[r]  = cv;
                hD[r] = og * fast_tanh(cv);
            }
            // scatter h tile (D layout) to row-major f16 LDS
            const int col = half * 16 + j;
            #pragma unroll
            for (int r = 0; r < 8; ++r)
                hbuf[(r + 8 * hi) * HID + col] = (_Float16)hD[r];
        }

        asm volatile("s_wait_dscnt 0" ::: "memory");

        // gather h in A layout: lane j = row j; lo lanes K 0..7,16..23, hi lanes +8
        {
            const int o1 = j * HID + hi * 8;
            v8h p0 = *(const v8h*)&hbuf[o1];
            v8h p1 = *(const v8h*)&hbuf[o1 + 16];
            hA = __builtin_shufflevector(p0, p1,
                 0,1,2,3,4,5,6,7,8,9,10,11,12,13,14,15);
        }

        // end of a scan: FC head + window rotation [d2, d1, out]
        if (slot == SEQ - 1) {
            const int f = step / SEQ;
            v8f acc;
            #pragma unroll
            for (int r = 0; r < 8; ++r) acc[r] = fcbias;
            v8f o = __builtin_amdgcn_wmma_f32_16x16x32_f16(
                false, hA, false, Bfc, (short)0, acc, false, false);

            // slot0 <- old slot2 (slot1 unchanged)
            if (lane < ROWS_PER_WAVE) {
                xbuf[0][lane][0] = xbuf[2][lane][0];
                xbuf[0][lane][1] = xbuf[2][lane][1];
            }
            asm volatile("s_wait_dscnt 0" ::: "memory");
            // slot2 <- out; also store to global [f][b][n]
            if (j < NCLS) {
                #pragma unroll
                for (int r = 0; r < 8; ++r) {
                    const int row = r + 8 * hi;
                    xbuf[2][row][j] = o[r];
                    out[((size_t)f * BATCH + b0 + row) * NCLS + j] = o[r];
                }
            }
            asm volatile("s_wait_dscnt 0" ::: "memory");
        }
    }
}

extern "C" void kernel_launch(void* const* d_in, const int* in_sizes, int n_in,
                              void* d_out, int out_size, void* d_ws, size_t ws_size,
                              hipStream_t stream) {
    const float* x    = (const float*)d_in[0];
    const float* W_ih = (const float*)d_in[1];
    const float* W_hh = (const float*)d_in[2];
    const float* b_ih = (const float*)d_in[3];
    const float* b_hh = (const float*)d_in[4];
    const float* W_fc = (const float*)d_in[5];
    const float* b_fc = (const float*)d_in[6];
    float* out = (float*)d_out;

    const int rows_per_block = ROWS_PER_WAVE * WAVES_PER_BLOCK;  // 128
    const int grid = BATCH / rows_per_block;                     // 4096
    lstm_wmma_kernel<<<grid, 256, 0, stream>>>(x, W_ih, W_hh, b_ih, b_hh,
                                               W_fc, b_fc, out);
}
